// GraphMoleculeModel_86723979640836
// MI455X (gfx1250) — compile-verified, matrix operand order checked by reading, not run on hardware
//
#include <hip/hip_runtime.h>
#include <hip/hip_bf16.h>
#include <cstdint>

typedef __attribute__((ext_vector_type(16))) _Float16 v16h;
typedef __attribute__((ext_vector_type(8)))  _Float16 v8h;
typedef __attribute__((ext_vector_type(8)))  float    v8f;

#define NN 60000
#define EE 200000
#define DD 128
#define HH 4
#define CC 128
#define HC 512      // H * C
#define LL 3
#define VAv 128
#define VBv 16

// ---------- order-preserving float <-> uint encoding for atomic max ----------
__device__ __forceinline__ unsigned enc_f32(float f) {
  unsigned b = __float_as_uint(f);
  return (b & 0x80000000u) ? ~b : (b | 0x80000000u);
}
__device__ __forceinline__ float dec_f32(unsigned k) {
  unsigned b = (k & 0x80000000u) ? (k & 0x7FFFFFFFu) : ~k;
  return __uint_as_float(b);
}

// A/B fragment for v_wmma_f32_16x16x32_f16: lane<16 holds K {k..k+7, k+16..k+23},
// lane>=16 holds K {k+8..k+15, k+24..k+31}; caller pre-adds koff = (lane>>4)*8.
__device__ __forceinline__ v16h load_frag(const _Float16* __restrict__ p) {
  v8h lo = *(const v8h*)(p);
  v8h hi = *(const v8h*)(p + 16);
  v16h r;
#pragma unroll
  for (int i = 0; i < 8; ++i) { r[i] = lo[i]; r[i + 8] = hi[i]; }
  return r;
}

// ---------- utility kernels ----------
__global__ void k_fill_zero(float* p, long n) {
  long i = (long)blockIdx.x * blockDim.x + threadIdx.x;
  if (i < n) p[i] = 0.0f;
}

__global__ void k_gather_emb(const int* __restrict__ x, const float* __restrict__ emb,
                             float* __restrict__ h, long n) {
  long i = (long)blockIdx.x * blockDim.x + threadIdx.x;
  if (i >= n) return;
  int node = (int)(i >> 7);
  int d = (int)(i & 127);
  h[i] = emb[(size_t)x[node] * DD + d];
}

__global__ void k_cast_f16(const float* __restrict__ src, _Float16* __restrict__ dst, long n) {
  long i = (long)blockIdx.x * blockDim.x + threadIdx.x;
  if (i < n) dst[i] = (_Float16)src[i];
}

// W[K][Nw] (f32, row-major) -> Wt[Nw][K] (f16) so GEMM B-columns are contiguous.
__global__ void k_transpose_f16(const float* __restrict__ W, _Float16* __restrict__ Wt,
                                int K, int Nw) {
  long i = (long)blockIdx.x * blockDim.x + threadIdx.x;
  if (i >= (long)K * Nw) return;
  int nn = (int)(i / K);
  int kk = (int)(i % K);
  Wt[i] = (_Float16)W[(size_t)kk * Nw + nn];
}

// ---------- WMMA GEMM: C[M][Nw] = A[M][128] * B[128][Nw] (+bias[Nw]) ----------
// One wave computes a 16x64 strip: A fragment is loaded once per k-step and
// reused across 4 B fragments (4x less A traffic, 16 wmma per K loop).
// A: f16 row-major [M][128]; Bt: f16 [Nw][128] (pre-transposed).
__global__ void k_gemm_wmma4(const _Float16* __restrict__ A, const _Float16* __restrict__ Bt,
                             const float* __restrict__ bias, float* __restrict__ Cmat,
                             int M, int Nw) {
  const int K = 128;
  int wid  = (int)((blockIdx.x * blockDim.x + threadIdx.x) >> 5);
  int lane = threadIdx.x & 31;
  int strips = Nw >> 6;                       // 64-wide output strips
  int total  = (M >> 4) * strips;
  if (wid >= total) return;
  int tm = wid / strips, tn = wid % strips;
  int m0 = tm << 4, n0 = tn << 6;
  int row  = lane & 15;
  int hi   = lane >> 4;
  int koff = hi << 3;

  const _Float16* pa = A  + (size_t)(m0 + row) * K + koff;
  const _Float16* pb = Bt + (size_t)(n0 + row) * K + koff;

  v8f acc[4] = {};
#pragma unroll
  for (int k0 = 0; k0 < K; k0 += 32) {
    v16h av = load_frag(pa + k0);
#pragma unroll
    for (int t = 0; t < 4; ++t) {
      v16h bv = load_frag(pb + (size_t)(t * 16) * K + k0);
      acc[t] = __builtin_amdgcn_wmma_f32_16x16x32_f16(false, av, false, bv,
                                                      (short)0, acc[t], false, false);
    }
  }
#pragma unroll
  for (int t = 0; t < 4; ++t) {
    int col = n0 + t * 16 + row;
    float bv2 = bias ? bias[col] : 0.0f;
#pragma unroll
    for (int r = 0; r < 8; ++r) {
      int m = m0 + r + (hi << 3);
      Cmat[(size_t)m * Nw + col] = acc[t][r] + bv2;
    }
  }
}

// ---------- bond head via WMMA: 16 edges per wave, K=256 (concat src|dst) ----
// A rows gathered per lane from hh; B = Wbt f16 [16][256] pre-transposed.
__global__ void k_bond_wmma(const int* __restrict__ ei, const _Float16* __restrict__ hh,
                            const _Float16* __restrict__ Wbt, const float* __restrict__ bb,
                            float* __restrict__ out) {
  int wid  = (int)((blockIdx.x * blockDim.x + threadIdx.x) >> 5);
  int lane = threadIdx.x & 31;
  if (wid >= EE / 16) return;
  int e0   = wid << 4;
  int row  = lane & 15;
  int hi   = lane >> 4;
  int koff = hi << 3;
  int e    = e0 + row;
  int sn   = ei[e], dn = ei[EE + e];

  const _Float16* ps = hh + (size_t)sn * DD;   // K 0..127
  const _Float16* pd = hh + (size_t)dn * DD;   // K 128..255
  const _Float16* pw = Wbt + (size_t)row * 256 + koff;

  v8f acc = {};
#pragma unroll
  for (int k0 = 0; k0 < 256; k0 += 32) {
    const _Float16* pa = (k0 < 128) ? (ps + k0 + koff) : (pd + (k0 - 128) + koff);
    v16h av = load_frag(pa);
    v16h bv = load_frag(pw + k0);
    acc = __builtin_amdgcn_wmma_f32_16x16x32_f16(false, av, false, bv,
                                                 (short)0, acc, false, false);
  }
  float bbv = bb[row];
#pragma unroll
  for (int r = 0; r < 8; ++r) {
    int edge = e0 + r + (hi << 3);
    out[(size_t)edge * VBv + row] = acc[r] + bbv;
  }
}

// ---------- edge pass 1: attention logits + segment max (wave per edge) ----------
__global__ void k_edge_logits(const int* __restrict__ ei, const int* __restrict__ eattr,
                              const float* __restrict__ xl, const float* __restrict__ xr,
                              const float* __restrict__ btab, const float* __restrict__ att,
                              float* __restrict__ logits, unsigned* __restrict__ mx) {
  int e = (int)((blockIdx.x * blockDim.x + threadIdx.x) >> 5);
  int lane = threadIdx.x & 31;
  if (e >= EE) return;
  int src = ei[e], dst = ei[EE + e];
  const float* pl = xl + (size_t)src * HC;
  const float* pr = xr + (size_t)dst * HC;
  const float* pe = btab + (size_t)eattr[e] * HC;
  int base = lane << 4;                 // 16 channels per lane; 8 lanes per head
  float part = 0.0f;
#pragma unroll
  for (int j = 0; j < 16; ++j) {
    float m = pl[base + j] + pr[base + j] + pe[base + j];
    m = (m > 0.0f) ? m : 0.2f * m;      // leaky_relu(0.2)
    part += m * att[base + j];
  }
  part += __shfl_xor(part, 1);
  part += __shfl_xor(part, 2);
  part += __shfl_xor(part, 4);          // reduce within each head's 8 lanes
  if ((lane & 7) == 0) {
    int hd = lane >> 3;
    logits[(size_t)e * HH + hd] = part;
    atomicMax(&mx[(size_t)dst * HH + hd], enc_f32(part));
  }
}

// ---------- edge pass 2: exp(logit - max) + segment sum ----------
__global__ void k_edge_expsum(const int* __restrict__ ei, float* __restrict__ logits,
                              const unsigned* __restrict__ mx, float* __restrict__ s) {
  long idx = (long)blockIdx.x * blockDim.x + threadIdx.x;
  if (idx >= (long)EE * HH) return;
  int e = (int)(idx >> 2), hd = (int)(idx & 3);
  int dst = ei[EE + e];
  float a = __expf(logits[idx] - dec_f32(mx[(size_t)dst * HH + hd]));
  logits[idx] = a;
  atomicAdd(&s[(size_t)dst * HH + hd], a);
}

// ---------- edge pass 3: weighted scatter-add of xl[src] into agg[dst] ----------
__global__ void k_edge_agg(const int* __restrict__ ei, const float* __restrict__ logits,
                           const float* __restrict__ s, const float* __restrict__ xl,
                           float* __restrict__ agg) {
  int e = (int)((blockIdx.x * blockDim.x + threadIdx.x) >> 5);
  int lane = threadIdx.x & 31;
  if (e >= EE) return;
  int src = ei[e], dst = ei[EE + e];
  int hd = lane >> 3;
  float alpha = logits[(size_t)e * HH + hd] /
                (s[(size_t)dst * HH + hd] + 1e-16f);
  int base = lane << 4;
  const float* pl = xl + (size_t)src * HC + base;
  float* pa = agg + (size_t)dst * HC + base;
#pragma unroll
  for (int j = 0; j < 16; ++j) atomicAdd(&pa[j], alpha * pl[j]);
}

// ---------- node update: h += relu(mean_heads(agg) + bias) ----------
__global__ void k_node_update(const float* __restrict__ agg, const float* __restrict__ bias,
                              float* __restrict__ h) {
  long i = (long)blockIdx.x * blockDim.x + threadIdx.x;
  if (i >= (long)NN * DD) return;
  int n = (int)(i >> 7), c = (int)(i & 127);
  const float* pa = agg + (size_t)n * HC;
  float u = 0.25f * (pa[c] + pa[c + 128] + pa[c + 256] + pa[c + 384]) + bias[c];
  h[i] += (u > 0.0f) ? u : 0.0f;
}

// ---------- host launcher ----------
static inline int nblk(long n, int t) { return (int)((n + t - 1) / t); }

extern "C" void kernel_launch(void* const* d_in, const int* in_sizes, int n_in,
                              void* d_out, int out_size, void* d_ws, size_t ws_size,
                              hipStream_t stream) {
  (void)in_sizes; (void)n_in; (void)out_size; (void)ws_size;
  const int*   x        = (const int*)  d_in[0];
  const int*   ei       = (const int*)  d_in[1];   // [2][E]
  const int*   eattr    = (const int*)  d_in[2];
  const float* atom_emb = (const float*)d_in[3];
  const float* bond_emb = (const float*)d_in[4];
  const float* Wl       = (const float*)d_in[5];   // [L][128][512]
  const float* bl       = (const float*)d_in[6];   // [L][512]
  const float* Wr       = (const float*)d_in[7];
  const float* br       = (const float*)d_in[8];
  const float* We       = (const float*)d_in[9];
  const float* att      = (const float*)d_in[10];  // [L][4][128]
  const float* bias     = (const float*)d_in[11];  // [L][128]
  const float* Wa       = (const float*)d_in[12];  // [128][128]
  const float* ba       = (const float*)d_in[13];
  const float* Wb       = (const float*)d_in[14];  // [256][16]
  const float* bb       = (const float*)d_in[15];

  char* ws = (char*)d_ws;
  size_t off = 0;
  auto take = [&](size_t bytes) -> void* {
    void* p = ws + off;
    off = (off + bytes + 255) & ~(size_t)255;
    return p;
  };
  float*     h      = (float*)    take((size_t)NN * DD * 4);
  _Float16*  hh     = (_Float16*) take((size_t)NN * DD * 2);
  float*     xl     = (float*)    take((size_t)NN * HC * 4);
  float*     xr     = (float*)    take((size_t)NN * HC * 4);
  float*     agg    = (float*)    take((size_t)NN * HC * 4);
  float*     logits = (float*)    take((size_t)EE * HH * 4);
  unsigned*  mx     = (unsigned*) take((size_t)NN * HH * 4);
  float*     ssum   = (float*)    take((size_t)NN * HH * 4);
  _Float16*  bembh  = (_Float16*) take((size_t)VBv * DD * 2);
  float*     btab   = (float*)    take((size_t)VBv * HC * 4);
  _Float16*  Wlt    = (_Float16*) take((size_t)LL * HC * DD * 2);
  _Float16*  Wrt    = (_Float16*) take((size_t)LL * HC * DD * 2);
  _Float16*  Wet    = (_Float16*) take((size_t)LL * HC * DD * 2);
  _Float16*  Wat    = (_Float16*) take((size_t)VAv * DD * 2);
  _Float16*  Wbt    = (_Float16*) take((size_t)VBv * 2 * DD * 2);

  // ---- one-time weight prep (deterministic, re-done every call) ----
  k_cast_f16<<<nblk((long)VBv * DD, 256), 256, 0, stream>>>(bond_emb, bembh, (long)VBv * DD);
  for (int l = 0; l < LL; ++l) {
    k_transpose_f16<<<nblk((long)DD * HC, 256), 256, 0, stream>>>(Wl + (size_t)l * DD * HC, Wlt + (size_t)l * HC * DD, DD, HC);
    k_transpose_f16<<<nblk((long)DD * HC, 256), 256, 0, stream>>>(Wr + (size_t)l * DD * HC, Wrt + (size_t)l * HC * DD, DD, HC);
    k_transpose_f16<<<nblk((long)DD * HC, 256), 256, 0, stream>>>(We + (size_t)l * DD * HC, Wet + (size_t)l * HC * DD, DD, HC);
  }
  k_transpose_f16<<<nblk((long)DD * VAv, 256), 256, 0, stream>>>(Wa, Wat, DD, VAv);
  k_transpose_f16<<<nblk((long)2 * DD * VBv, 256), 256, 0, stream>>>(Wb, Wbt, 2 * DD, VBv);

  // ---- initial node features ----
  k_gather_emb<<<nblk((long)NN * DD, 256), 256, 0, stream>>>(x, atom_emb, h, (long)NN * DD);

  // ---- GATv2 layers ----
  for (int l = 0; l < LL; ++l) {
    k_cast_f16<<<nblk((long)NN * DD, 256), 256, 0, stream>>>(h, hh, (long)NN * DD);

    long gWaves = (long)(NN / 16) * (HC / 64);
    k_gemm_wmma4<<<nblk(gWaves * 32, 256), 256, 0, stream>>>(hh, Wlt + (size_t)l * HC * DD, bl + (size_t)l * HC, xl, NN, HC);
    k_gemm_wmma4<<<nblk(gWaves * 32, 256), 256, 0, stream>>>(hh, Wrt + (size_t)l * HC * DD, br + (size_t)l * HC, xr, NN, HC);
    long tWaves = (long)(VBv / 16) * (HC / 64);
    k_gemm_wmma4<<<nblk(tWaves * 32, 256), 256, 0, stream>>>(bembh, Wet + (size_t)l * HC * DD, nullptr, btab, VBv, HC);

    k_fill_zero<<<nblk((long)NN * HH, 256), 256, 0, stream>>>((float*)mx, (long)NN * HH);
    k_fill_zero<<<nblk((long)NN * HH, 256), 256, 0, stream>>>(ssum, (long)NN * HH);
    k_fill_zero<<<nblk((long)NN * HC, 256), 256, 0, stream>>>(agg, (long)NN * HC);

    k_edge_logits<<<nblk((long)EE * 32, 256), 256, 0, stream>>>(ei, eattr, xl, xr, btab, att + (size_t)l * HH * CC, logits, mx);
    k_edge_expsum<<<nblk((long)EE * HH, 256), 256, 0, stream>>>(ei, logits, mx, ssum);
    k_edge_agg<<<nblk((long)EE * 32, 256), 256, 0, stream>>>(ei, logits, ssum, xl, agg);
    k_node_update<<<nblk((long)NN * DD, 256), 256, 0, stream>>>(agg, bias + (size_t)l * CC, h);
  }

  // ---- output heads ----
  float* out_atom = (float*)d_out;                       // [N][128]
  float* out_bond = out_atom + (size_t)NN * VAv;         // [E][16]

  k_cast_f16<<<nblk((long)NN * DD, 256), 256, 0, stream>>>(h, hh, (long)NN * DD);
  long aWaves = (long)(NN / 16) * (VAv / 64);
  k_gemm_wmma4<<<nblk(aWaves * 32, 256), 256, 0, stream>>>(hh, Wat, ba, out_atom, NN, VAv);
  k_bond_wmma<<<nblk((long)(EE / 16) * 32, 256), 256, 0, stream>>>(ei, hh, Wbt, bb, out_bond);
}
